// OneNetFusionHead_26259430048585
// MI455X (gfx1250) — matrix operand order repaired and checked by reading, not compiled, not run on hardware
//
#include <hip/hip_runtime.h>
#include <hip/hip_bf16.h>
#include <stdint.h>

typedef __attribute__((ext_vector_type(16))) _Float16 v16h;
typedef __attribute__((ext_vector_type(8)))  _Float16 v8h;
typedef __attribute__((ext_vector_type(8)))  float    v8f;
typedef __attribute__((ext_vector_type(4)))  unsigned int v4u;
typedef __attribute__((ext_vector_type(8)))  int      v8i;
typedef __attribute__((ext_vector_type(4)))  int      v4i;

#if __has_builtin(__builtin_amdgcn_tensor_load_to_lds) && __has_builtin(__builtin_amdgcn_s_wait_tensorcnt)
#define HAVE_TDM 1
#else
#define HAVE_TDM 0
#endif

#define H_   128
#define W_   128
#define HW_  16384
#define B_   2
#define BHW_ 32768   // B_*HW_
#define HP_  130     // padded height (1-px halo)
#define WP_  136     // padded width  (halo + alignment pad)
#define CHP_ (HP_*WP_)

// padded f16 tensor index for (b, c, pixel p) with C channels
__device__ __forceinline__ size_t pidx(int b, int C, int c, int p)
{
    int y = p >> 7, x = p & 127;
    return ((size_t)(b * C + c) * HP_ + y + 1) * WP_ + x + 1;
}

// ============================================================================
// Implicit-GEMM conv via WMMA, f16 in / f32 accumulate (SAME pad, stride 1).
// x: zero-padded (B,C,130,136) f16.  wgt: TAP-MAJOR (Cout, KSZ*Cin) f16.
// Weights staged by the Tensor Data Mover (one 2D D# per K-step) when the
// builtin exists; activations staged with vector loads. Double-buffered LDS,
// one barrier per K-step.
// ============================================================================
template<int KH, int KW>
__launch_bounds__(256)
__global__ void wmma_conv_kernel(const _Float16* __restrict__ x,
                                 const _Float16* __restrict__ wgt,
                                 const float* __restrict__ bias,
                                 float* __restrict__ out,            // (B,Cout,HW)
                                 _Float16* __restrict__ out16,       // padded or null
                                 int Cin, int Cout, int relu)
{
    constexpr int KSZ = KH * KW;
    constexpr int RO  = 1 - KH / 2;
    constexpr int CO  = 1 - KW / 2;
    __shared__ __align__(16) _Float16 sA[2][128][40];   // [buf][pixel][k]
    __shared__ __align__(16) _Float16 sW[2][64][40];    // [buf][chan ][k]

    const int tid     = threadIdx.x;
    const int lane    = tid & 31;
    const int wid     = tid >> 5;
    const int chTile  = wid & 3;
    const int pixSlab = wid >> 2;
    const int K       = Cin * KSZ;     // multiple of 32
    const int pBase   = blockIdx.x * 128;
    const int coBase  = blockIdx.y * 64;

    // ---- per-thread A-staging invariants ----------------------------------
    const int octA   = tid & 15;
    const int kkT    = tid >> 4;
    const int m0     = octA * 8;
    const int pgA    = pBase + m0;
    const int bA     = pgA >> 14;
    const int ppA    = pgA & (HW_ - 1);
    const int pixOff = (ppA >> 7) * WP_ + (ppA & 127);
    const size_t aBase = (size_t)bA * Cin * CHP_;

    const int nW   = tid >> 2;
    const int octW = tid & 3;
    const int coW  = coBase + nW;
    const bool coOK = coW < Cout;
    const _Float16* wBase = wgt + (size_t)coW * K + octW * 8;

    v8f acc[4] = {};
    union FragH { v16h h; uint32_t u[8]; };
    const int half = lane >> 4;
    const int l16  = lane & 15;

    auto stageA = [&](int buf, int cBase, int tap) {
        const int ky = tap / KW, kx = tap - ky * KW;      // scalar-uniform
        const int stepOff = (ky + RO) * WP_ + kx + CO;
        #pragma unroll
        for (int i = 0; i < 2; ++i) {
            int kk = i * 16 + kkT;
            v8h av;
            __builtin_memcpy(&av,
                &x[aBase + (size_t)(cBase + kk) * CHP_ + pixOff + stepOff],
                sizeof(av));
            #pragma unroll
            for (int j = 0; j < 8; ++j) sA[buf][m0 + j][kk] = av[j];
        }
    };

#if HAVE_TDM
    // TDM: DMA one 64x32 f16 weight tile (rows=out-chans, stride K) into
    // sW[buf] with hardware LDS padding 16DW data + 4DW pad = our [64][40].
    const int rowsValid = Cout - coBase;                 // > 0
    auto stageW_tdm = [&](int buf, int k0) {
        uint64_t gaddr = (uint64_t)(uintptr_t)(wgt + (size_t)coBase * K + k0);
        uint32_t laddr = (uint32_t)(uintptr_t)&sW[buf][0][0];
        v4u g0;
        g0[0] = 1u;                                      // count=1 (valid D#)
        g0[1] = laddr;                                   // lds_addr
        g0[2] = (uint32_t)gaddr;                         // global_addr lo
        g0[3] = (uint32_t)(gaddr >> 32) | 0x80000000u;   // addr hi | type=2
        v8i g1;
        g1[0] = (int)((1u << 16)        // data_size = 2 bytes
                    | (1u << 20)        // pad_enable
                    | (3u << 22)        // pad_interval: every 16 DWORDs
                    | (3u << 25));      // pad_amount: 4 DWORDs
        g1[1] = (int)(32u << 16);                        // tensor_dim0 lo16 = 32
        g1[2] = (int)(((uint32_t)rowsValid & 0xFFFFu) << 16); // td0 hi=0 | td1 lo16
        g1[3] = (int)(((uint32_t)rowsValid >> 16) | (32u << 16)); // td1 hi | tile_dim0=32
        g1[4] = 64;                                      // tile_dim1=64, tile_dim2=0
        g1[5] = K;                                       // dim0 stride lo32 (elements)
        g1[6] = 0;
        g1[7] = 0;
        v4i gz4 = {};
        v8i gz8 = {};
        __builtin_amdgcn_tensor_load_to_lds(g0, g1, gz4, gz4, gz8, 0);
    };
#endif
    auto stageW_vec = [&](int buf, int k0) {
        v8h wv = {};
        if (coOK) wv = *(const v8h*)&wBase[k0];
        *(v8h*)&sW[buf][nW][octW * 8] = wv;
    };

    // ---- prologue ----------------------------------------------------------
#if HAVE_TDM
    if (wid == 0) stageW_tdm(0, 0);
    stageA(0, 0, 0);
    if (wid == 0) __builtin_amdgcn_s_wait_tensorcnt(0);
#else
    stageW_vec(0, 0);
    stageA(0, 0, 0);
#endif
    __syncthreads();

    int cB = 0, tap = 0, buf = 0;
    for (int k0 = 0; k0 < K; k0 += 32) {
        int cBn = cB + 32, tn = tap;
        if (cBn == Cin) { cBn = 0; tn = tap + 1; }
        if (k0 + 32 < K) {
#if HAVE_TDM
            if (wid == 0) stageW_tdm(buf ^ 1, k0 + 32);
            stageA(buf ^ 1, cBn, tn);
#else
            stageW_vec(buf ^ 1, k0 + 32);
            stageA(buf ^ 1, cBn, tn);
            if (lane == 0 && coOK) __builtin_prefetch(&wBase[k0 + 64], 0, 1);
#endif
        }

        // A fragment (weights 16x32)
        FragH fa;
        {
            int row = chTile * 16 + l16;
            #pragma unroll
            for (int j = 0; j < 4; ++j)
                fa.u[j]     = *(const uint32_t*)&sW[buf][row][half * 8 + 2 * j];
            #pragma unroll
            for (int j = 0; j < 4; ++j)
                fa.u[4 + j] = *(const uint32_t*)&sW[buf][row][16 + half * 8 + 2 * j];
        }
        // B fragments (activations 32x16)
        #pragma unroll
        for (int s = 0; s < 4; ++s) {
            FragH fb;
            int row = pixSlab * 64 + s * 16 + l16;
            #pragma unroll
            for (int j = 0; j < 4; ++j)
                fb.u[j]     = *(const uint32_t*)&sA[buf][row][half * 8 + 2 * j];
            #pragma unroll
            for (int j = 0; j < 4; ++j)
                fb.u[4 + j] = *(const uint32_t*)&sA[buf][row][16 + half * 8 + 2 * j];
            acc[s] = __builtin_amdgcn_wmma_f32_16x16x32_f16(false, fa.h, false, fb.h,
                                                            (short)0, acc[s], false, false);
        }
#if HAVE_TDM
        if (wid == 0) __builtin_amdgcn_s_wait_tensorcnt(0);
#endif
        __syncthreads();
        cB = cBn; tap = tn; buf ^= 1;
    }

    // ---- store -------------------------------------------------------------
    #pragma unroll
    for (int s = 0; s < 4; ++s) {
        #pragma unroll
        for (int r = 0; r < 8; ++r) {
            int ch = coBase + chTile * 16 + r + (half ? 8 : 0);
            if (ch >= Cout) continue;
            int pg = pBase + pixSlab * 64 + s * 16 + l16;
            int b = pg >> 14, pp = pg & (HW_ - 1);
            float v = acc[s][r] + (bias ? bias[ch] : 0.f);
            if (relu) v = fmaxf(v, 0.f);
            out[(size_t)(b * Cout + ch) * HW_ + pp] = v;
            if (out16) out16[pidx(b, Cout, ch, pp)] = (_Float16)v;
        }
    }
}

// ============================================================================
__global__ void zero_u32_kernel(uint32_t* __restrict__ p, size_t n)
{
    size_t i = (size_t)blockIdx.x * 256 + threadIdx.x;
    if (i < n) p[i] = 0u;
}

// weight f32 (Cout,Cin,KSZ) -> f16 tap-major (Cout, KSZ*Cin)
__global__ void cvt_w_kernel(const float* __restrict__ src,
                             _Float16* __restrict__ dst,
                             int Cout, int Cin, int KSZ)
{
    int idx = blockIdx.x * 256 + threadIdx.x;
    int total = Cout * Cin * KSZ;
    if (idx >= total) return;
    int co = idx / (Cin * KSZ);
    int r  = idx - co * (Cin * KSZ);
    int t  = r / Cin;
    int c  = r - t * Cin;
    dst[idx] = (_Float16)src[(size_t)(co * Cin + c) * KSZ + t];
}

// (B,C,H,W) f32 -> padded (B,C,130,136) f16 with zero halo
__global__ void pad_cvt_kernel(const float* __restrict__ src,
                               _Float16* __restrict__ dst, int C)
{
    int idx = blockIdx.x * 256 + threadIdx.x;
    int total = B_ * C * CHP_;
    if (idx >= total) return;
    int xp = idx % WP_;
    int r  = idx / WP_;
    int yp = r % HP_;
    int bc = r / HP_;
    _Float16 v = (_Float16)0.f;
    if (yp >= 1 && yp <= H_ && xp >= 1 && xp <= W_)
        v = (_Float16)src[(size_t)bc * HW_ + (yp - 1) * W_ + (xp - 1)];
    dst[idx] = v;
}

// per-channel mean + rsqrt(var+eps) over (B,H,W)
__global__ void bn_stats_kernel(const float* __restrict__ x,
                                float* __restrict__ stats, int C)
{
    __shared__ float ss[256], sq[256];
    int ch = blockIdx.x;
    float s = 0.f, q = 0.f;
    for (int i = threadIdx.x; i < BHW_; i += 256) {
        int b = i >> 14, p = i & (HW_ - 1);
        float v = x[(size_t)(b * C + ch) * HW_ + p];
        s += v; q += v * v;
    }
    ss[threadIdx.x] = s; sq[threadIdx.x] = q;
    __syncthreads();
    for (int off = 128; off > 0; off >>= 1) {
        if (threadIdx.x < off) {
            ss[threadIdx.x] += ss[threadIdx.x + off];
            sq[threadIdx.x] += sq[threadIdx.x + off];
        }
        __syncthreads();
    }
    if (threadIdx.x == 0) {
        float mean = ss[0] / (float)BHW_;
        float var  = sq[0] / (float)BHW_ - mean * mean;
        stats[ch * 2]     = mean;
        stats[ch * 2 + 1] = rsqrtf(var + 1e-5f);
    }
}

// in-place x = relu(g*(x-m)*istd + b); optional padded f16 mirror
__global__ void bn_apply_relu_kernel(float* __restrict__ x,
                                     _Float16* __restrict__ x16,
                                     const float* __restrict__ stats,
                                     const float* __restrict__ g,
                                     const float* __restrict__ b, int C)
{
    int idx = blockIdx.x * 256 + threadIdx.x;
    int total = B_ * C * HW_;
    if (idx >= total) return;
    int bb = idx / (C * HW_);
    int ch = (idx / HW_) % C;
    int p  = idx & (HW_ - 1);
    float v = g[ch] * (x[idx] - stats[ch * 2]) * stats[ch * 2 + 1] + b[ch];
    v = fmaxf(v, 0.f);
    x[idx] = v;
    if (x16) x16[pidx(bb, C, ch, p)] = (_Float16)v;
}

// copy src (B,Cs,HW) into dst (B,Cd,HW) at channel offset chOff
__global__ void copy_ch_kernel(const float* __restrict__ src,
                               float* __restrict__ dst,
                               int Cs, int Cd, int chOff)
{
    int idx = blockIdx.x * 256 + threadIdx.x;
    int total = B_ * Cs * HW_;
    if (idx >= total) return;
    int b = idx / (Cs * HW_);
    int r = idx - b * (Cs * HW_);
    int c = r / HW_, p = r - c * HW_;
    dst[(size_t)(b * Cd + chOff + c) * HW_ + p] = src[idx];
}

// ============================================================================
// Modulated deformable conv 3x3 (+bias,+relu). One thread per (b,pixel).
// ============================================================================
__launch_bounds__(128)
__global__ void mdcn_relu_kernel(const float* __restrict__ xb,
                                 const float* __restrict__ offr,
                                 const float* __restrict__ wgt,
                                 const float* __restrict__ bias,
                                 float* __restrict__ out,
                                 _Float16* __restrict__ out16,
                                 int C)
{
    int idx = blockIdx.x * 128 + threadIdx.x;
    if (idx >= BHW_) return;
    int b = idx >> 14, p = idx & (HW_ - 1);
    int y = p >> 7, x = p & 127;

    float acc[64];
    #pragma unroll
    for (int o = 0; o < 64; ++o) acc[o] = 0.f;

    const float* xbb  = xb + (size_t)b * C * HW_;
    const float* offb = offr + (size_t)b * 27 * HW_;

    for (int k = 0; k < 9; ++k) {
        float dy  = offb[(size_t)k * HW_ + p];
        float dx  = offb[(size_t)(9 + k) * HW_ + p];
        float mr  = offb[(size_t)(18 + k) * HW_ + p];
        float msk = 1.f / (1.f + expf(-mr));
        float py = (float)y + (float)(k / 3) - 1.f + dy;
        float px = (float)x + (float)(k % 3) - 1.f + dx;
        float y0f = floorf(py), x0f = floorf(px);
        float wy = py - y0f, wx = px - x0f;
        int iy0 = (int)y0f, ix0 = (int)x0f;

        float cw[4]; int cidx[4];
        #pragma unroll
        for (int q = 0; q < 4; ++q) {
            int dyi = q >> 1, dxi = q & 1;
            int yy = iy0 + dyi, xx = ix0 + dxi;
            bool valid = (yy >= 0 && yy < H_ && xx >= 0 && xx < W_);
            int yc = yy < 0 ? 0 : (yy > H_ - 1 ? H_ - 1 : yy);
            int xc = xx < 0 ? 0 : (xx > W_ - 1 ? W_ - 1 : xx);
            cidx[q] = yc * W_ + xc;
            float w2 = (dyi ? wy : 1.f - wy) * (dxi ? wx : 1.f - wx);
            cw[q] = valid ? w2 : 0.f;
        }
        for (int c = 0; c < C; ++c) {
            const float* xc = xbb + (size_t)c * HW_;
            float v = cw[0] * xc[cidx[0]] + cw[1] * xc[cidx[1]]
                    + cw[2] * xc[cidx[2]] + cw[3] * xc[cidx[3]];
            v *= msk;
            const float* wp = wgt + (size_t)c * 9 + k;
            #pragma unroll 8
            for (int o = 0; o < 64; ++o) acc[o] += v * wp[(size_t)o * C * 9];
        }
    }
    for (int o = 0; o < 64; ++o) {
        float v = fmaxf(acc[o] + bias[o], 0.f);
        out[(size_t)(b * 64 + o) * HW_ + p] = v;
        out16[pidx(b, 64, o, p)] = (_Float16)v;
    }
}

// ============================================================================
// Attention softmax over 3 branches -> final channels 512..575 (f32 + f16)
// ============================================================================
__global__ void fusion_kernel(const float* __restrict__ q,
                              const float* __restrict__ kc,
                              const float* __restrict__ kn,
                              const float* __restrict__ kf,
                              const float* __restrict__ ctr,
                              const float* __restrict__ cnr,
                              const float* __restrict__ fg,
                              float* __restrict__ outFinal,
                              _Float16* __restrict__ outFinal16)
{
    int idx = blockIdx.x * 256 + threadIdx.x;
    if (idx >= BHW_) return;
    int b = idx >> 14, p = idx & (HW_ - 1);
    float a0 = 0.f, a1 = 0.f, a2 = 0.f;
    for (int c = 0; c < 16; ++c) {
        size_t o = (size_t)(b * 16 + c) * HW_ + p;
        float qv = q[o];
        a0 += qv * kc[o];  a1 += qv * kn[o];  a2 += qv * kf[o];
    }
    float m = fmaxf(a0, fmaxf(a1, a2));
    float e0 = expf(a0 - m), e1 = expf(a1 - m), e2 = expf(a2 - m);
    float inv = 1.f / (e0 + e1 + e2);
    e0 *= inv; e1 *= inv; e2 *= inv;
    for (int c = 0; c < 64; ++c) {
        size_t o = (size_t)(b * 64 + c) * HW_ + p;
        float v = e0 * ctr[o] + e1 * cnr[o] + e2 * fg[o];
        outFinal[((size_t)b * 576 + 512 + c) * HW_ + p] = v;
        outFinal16[pidx(b, 576, 512 + c, p)] = (_Float16)v;
    }
}

// x (exact fp32) -> final channels 0..511 in d_out, plus padded f16 mirror
__global__ void copy_x_to_final(const float* __restrict__ x,
                                float* __restrict__ outFinal,
                                _Float16* __restrict__ outFinal16)
{
    int idx = blockIdx.x * 256 + threadIdx.x;
    int total = B_ * 512 * HW_;
    if (idx >= total) return;
    int b = idx / (512 * HW_);
    int r = idx - b * (512 * HW_);
    int c = r / HW_, p = r & (HW_ - 1);
    float v = x[idx];
    outFinal[(size_t)b * 576 * HW_ + r] = v;
    outFinal16[pidx(b, 576, c, p)] = (_Float16)v;
}

// ============================================================================
extern "C" void kernel_launch(void* const* d_in, const int* in_sizes, int n_in,
                              void* d_out, int out_size, void* d_ws, size_t ws_size,
                              hipStream_t stream)
{
    const float* X = (const float*)d_in[0];
    auto in = [&](int i) { return (const float*)d_in[i]; };

    float* out = (float*)d_out;
    float* OUT_CM    = out;
    float* OUT_NM    = out + (size_t)3   * BHW_;
    float* OUT_FM    = out + (size_t)15  * BHW_;
    float* OUT_FINAL = out + (size_t)18  * BHW_;
    float* OUT_CLS   = out + (size_t)594 * BHW_;
    float* OUT_BBOX  = out + (size_t)597 * BHW_;

    // ---- workspace carving (16B-aligned slots) -----------------------------
    float* ws = (float*)d_ws;
    size_t off = 0;
    auto allocF = [&](size_t nfl) { float* p = ws + off;
                                    off += (nfl + 3) & ~(size_t)3; return p; };
    auto allocH = [&](size_t nh)  { _Float16* p = (_Float16*)(ws + off);
                                    off += ((nh + 1) / 2 + 3) & ~(size_t)3; return p; };
    float* cf     = allocF((size_t)64 * BHW_);
    float* nf     = allocF((size_t)64 * BHW_);
    float* ff     = allocF((size_t)64 * BHW_);
    float* ctr_in = allocF((size_t)67 * BHW_);
    float* cnr_in = allocF((size_t)76 * BHW_);
    float* fg_in  = allocF((size_t)67 * BHW_);
    float* offc   = allocF((size_t)27 * BHW_);
    float* offn   = allocF((size_t)27 * BHW_);
    float* offf   = allocF((size_t)27 * BHW_);
    float* ctr    = allocF((size_t)64 * BHW_);
    float* cnr    = allocF((size_t)64 * BHW_);
    float* fgb    = allocF((size_t)64 * BHW_);
    float* qb     = allocF((size_t)16 * BHW_);
    float* kc     = allocF((size_t)16 * BHW_);
    float* kn     = allocF((size_t)16 * BHW_);
    float* kf     = allocF((size_t)16 * BHW_);
    float* h1     = allocF((size_t)64 * BHW_);
    float* h2     = allocF((size_t)64 * BHW_);
    float* st     = allocF((size_t)7 * 128);
    float *st_cf = st, *st_nf = st + 128, *st_ff = st + 256, *st_q = st + 384;
    float *st_kc = st + 512, *st_kn = st + 640, *st_kf = st + 768;
    size_t arenaStart = off;
    _Float16* Xhp    = allocH((size_t)B_ * 512 * CHP_);
    _Float16* cfh    = allocH((size_t)B_ * 64 * CHP_);
    _Float16* nfh    = allocH((size_t)B_ * 64 * CHP_);
    _Float16* ffh    = allocH((size_t)B_ * 64 * CHP_);
    _Float16* ctrh   = allocH((size_t)B_ * 64 * CHP_);
    _Float16* cnrh   = allocH((size_t)B_ * 64 * CHP_);
    _Float16* fgbh   = allocH((size_t)B_ * 64 * CHP_);
    _Float16* finalh = allocH((size_t)B_ * 576 * CHP_);
    _Float16* h1h    = allocH((size_t)B_ * 64 * CHP_);
    _Float16* h2h    = allocH((size_t)B_ * 64 * CHP_);
    size_t arenaU32 = off - arenaStart;
    _Float16* wcc = allocH(294912);  _Float16* wch = allocH(1728);
    _Float16* wnc = allocH(294912);  _Float16* wnh = allocH(6912);
    _Float16* wfc = allocH(294912);  _Float16* wfh = allocH(1728);
    _Float16* woc = allocH(13824);   _Float16* won = allocH(13824);
    _Float16* wof = allocH(13824);
    _Float16* wbq = allocH(8192);
    _Float16* wkc = allocH(1024);    _Float16* wkn = allocH(1024);
    _Float16* wkf = allocH(1024);
    _Float16* wc1 = allocH(331776);  _Float16* wc2 = allocH(1728);
    _Float16* wb1 = allocH(331776);  _Float16* wb2 = allocH(4608);
    (void)ws_size; (void)n_in; (void)in_sizes; (void)out_size;

    const dim3 blk(256);
    auto cvtw = [&](const float* s, _Float16* d, int Cout, int Cin, int KSZ) {
        int n = Cout * Cin * KSZ;
        cvt_w_kernel<<<dim3((n + 255) / 256), blk, 0, stream>>>(s, d, Cout, Cin, KSZ);
    };
    auto conv = [&](const _Float16* x, const _Float16* w, const float* b,
                    float* o, _Float16* o16, int Cin, int Cout, int k3, int relu) {
        dim3 grid(BHW_ / 128, (Cout + 63) / 64);
        if (k3)
            wmma_conv_kernel<3,3><<<grid, blk, 0, stream>>>(x, w, b, o, o16, Cin, Cout, relu);
        else
            wmma_conv_kernel<1,1><<<grid, blk, 0, stream>>>(x, w, b, o, o16, Cin, Cout, relu);
    };
    auto bn = [&](float* x, _Float16* x16, float* stats,
                  const float* g, const float* b, int C) {
        bn_stats_kernel<<<dim3(C), blk, 0, stream>>>(x, stats, C);
        int tot = B_ * C * HW_;
        bn_apply_relu_kernel<<<dim3((tot + 255) / 256), blk, 0, stream>>>(x, x16, stats, g, b, C);
    };
    auto cpych = [&](const float* s, float* d, int Cs, int Cd, int co) {
        int tot = B_ * Cs * HW_;
        copy_ch_kernel<<<dim3((tot + 255) / 256), blk, 0, stream>>>(s, d, Cs, Cd, co);
    };

    zero_u32_kernel<<<dim3((unsigned)((arenaU32 + 255) / 256)), blk, 0, stream>>>(
        (uint32_t*)(ws + arenaStart), arenaU32);

    {
        int tot = B_ * 512 * CHP_;
        pad_cvt_kernel<<<dim3((tot + 255) / 256), blk, 0, stream>>>(X, Xhp, 512);
    }
    cvtw(in(1),  wcc, 64, 512, 9);  cvtw(in(5),  wch, 3,  64, 9);
    cvtw(in(7),  wnc, 64, 512, 9);  cvtw(in(11), wnh, 12, 64, 9);
    cvtw(in(13), wfc, 64, 512, 9);  cvtw(in(17), wfh, 3,  64, 9);
    cvtw(in(19), woc, 27, 512, 1);  cvtw(in(23), won, 27, 512, 1);
    cvtw(in(27), wof, 27, 512, 1);
    cvtw(in(31), wbq, 16, 512, 1);
    cvtw(in(35), wkc, 16, 64, 1);   cvtw(in(39), wkn, 16, 64, 1);
    cvtw(in(43), wkf, 16, 64, 1);
    cvtw(in(47), wc1, 64, 576, 9);  cvtw(in(49), wc2, 3, 64, 9);
    cvtw(in(51), wb1, 64, 576, 9);  cvtw(in(53), wb2, 8, 64, 9);

    conv(Xhp, wcc, in(2),  cf, nullptr, 512, 64, 1, 0);  bn(cf, cfh, st_cf, in(3),  in(4),  64);
    conv(Xhp, wnc, in(8),  nf, nullptr, 512, 64, 1, 0);  bn(nf, nfh, st_nf, in(9),  in(10), 64);
    conv(Xhp, wfc, in(14), ff, nullptr, 512, 64, 1, 0);  bn(ff, ffh, st_ff, in(15), in(16), 64);

    conv(cfh, wch, in(6),  OUT_CM, nullptr, 64, 3,  1, 0);
    conv(nfh, wnh, in(12), OUT_NM, nullptr, 64, 12, 1, 0);
    conv(ffh, wfh, in(18), OUT_FM, nullptr, 64, 3,  1, 0);

    cpych(cf, ctr_in, 64, 67, 0);  cpych(OUT_CM, ctr_in, 3,  67, 64);
    cpych(nf, cnr_in, 64, 76, 0);  cpych(OUT_NM, cnr_in, 12, 76, 64);
    cpych(ff, fg_in,  64, 67, 0);  cpych(OUT_FM, fg_in,  3,  67, 64);

    conv(Xhp, woc, in(20), offc, nullptr, 512, 27, 0, 0);
    conv(Xhp, won, in(24), offn, nullptr, 512, 27, 0, 0);
    conv(Xhp, wof, in(28), offf, nullptr, 512, 27, 0, 0);

    dim3 mg((BHW_ + 127) / 128);
    mdcn_relu_kernel<<<mg, dim3(128), 0, stream>>>(ctr_in, offc, in(21), in(22), ctr, ctrh, 67);
    mdcn_relu_kernel<<<mg, dim3(128), 0, stream>>>(cnr_in, offn, in(25), in(26), cnr, cnrh, 76);
    mdcn_relu_kernel<<<mg, dim3(128), 0, stream>>>(fg_in,  offf, in(29), in(30), fgb, fgbh, 67);

    conv(Xhp,  wbq, in(32), qb, nullptr, 512, 16, 0, 0);  bn(qb, nullptr, st_q,  in(33), in(34), 16);
    conv(ctrh, wkc, in(36), kc, nullptr, 64,  16, 0, 0);  bn(kc, nullptr, st_kc, in(37), in(38), 16);
    conv(cnrh, wkn, in(40), kn, nullptr, 64,  16, 0, 0);  bn(kn, nullptr, st_kn, in(41), in(42), 16);
    conv(fgbh, wkf, in(44), kf, nullptr, 64,  16, 0, 0);  bn(kf, nullptr, st_kf, in(45), in(46), 16);

    fusion_kernel<<<dim3((BHW_ + 255) / 256), blk, 0, stream>>>(
        qb, kc, kn, kf, ctr, cnr, fgb, OUT_FINAL, finalh);
    {
        int tot = B_ * 512 * HW_;
        copy_x_to_final<<<dim3((tot + 255) / 256), blk, 0, stream>>>(X, OUT_FINAL, finalh);
    }

    conv(finalh, wc1, in(48), h1, h1h, 576, 64, 1, 1);
    conv(h1h,    wc2, in(50), OUT_CLS, nullptr, 64, 3, 1, 0);
    conv(finalh, wb1, in(52), h2, h2h, 576, 64, 1, 1);
    conv(h2h,    wb2, in(54), OUT_BBOX, nullptr, 64, 8, 1, 0);
}
